// CNBlock_20444044329481
// MI455X (gfx1250) — matrix operand order repaired and checked by reading, not compiled, not run on hardware
//
#include <hip/hip_runtime.h>
#include <stdint.h>

typedef __attribute__((ext_vector_type(16))) __bf16 v16bf;
typedef __attribute__((ext_vector_type(8)))  float  v8f;

// Problem sizes
constexpr int Bn  = 32, Cn = 256, Hn = 56, Wn = 56, HIDn = 1024;
constexpr int ROWS = Bn * Hn * Wn;          // 100352
constexpr int RT   = 32;                    // rows per GEMM workgroup (83KB LDS -> 3 WGs/WGP)
constexpr int NBLK_GEMM = ROWS / RT;        // 3136

// LDS layout for fused MLP kernel (byte strides padded +16B to spread banks)
constexpr int A_STRIDE = Cn   * 2 + 16;     // 528
constexpr int H_STRIDE = HIDn * 2 + 16;     // 2064
constexpr int A_LDS_BYTES = RT * A_STRIDE;  // 16896
constexpr int H_LDS_BYTES = RT * H_STRIDE;  // 66048
constexpr int GEMM_LDS    = A_LDS_BYTES + H_LDS_BYTES; // 82944 -> 3 blocks/WGP

// Workspace layout (bytes)
constexpr size_t WS_A   = 0;                                    // bf16 A [ROWS][256]
constexpr size_t WS_W1T = (size_t)ROWS * Cn * 2;                // 51,380,224
constexpr size_t WS_W2T = WS_W1T + (size_t)HIDn * Cn * 2;       // + 524,288

__device__ __forceinline__ uint16_t f2bf(float f) {
    uint32_t u = __float_as_uint(f);
    u += 0x7FFFu + ((u >> 16) & 1u);        // round-to-nearest-even
    return (uint16_t)(u >> 16);
}

union BF16x16 { uint32_t u[8]; v16bf v; };

__device__ __forceinline__ v16bf frag_from(const uint4 lo, const uint4 hi) {
    BF16x16 t;
    t.u[0] = lo.x; t.u[1] = lo.y; t.u[2] = lo.z; t.u[3] = lo.w;
    t.u[4] = hi.x; t.u[5] = hi.y; t.u[6] = hi.z; t.u[7] = hi.w;
    return t.v;
}

__device__ __forceinline__ v8f wmma_bf16(v16bf a, v16bf b, v8f c) {
    return __builtin_amdgcn_wmma_f32_16x16x32_bf16(
        false, a, false, b, (short)0, c, false, false);
}

// ---------------------------------------------------------------------------
// Weight transpose + bf16 conversion: w1[c][f] -> w1t[f][c], w2[f][c] -> w2t[c][f]
// ---------------------------------------------------------------------------
__global__ void prep_w1t_kernel(const float* __restrict__ w1, uint16_t* __restrict__ w1t) {
    int idx = blockIdx.x * blockDim.x + threadIdx.x;       // idx over 256*1024
    int c = idx >> 10, f = idx & 1023;
    w1t[f * Cn + c] = f2bf(w1[c * HIDn + f]);
}
__global__ void prep_w2t_kernel(const float* __restrict__ w2, uint16_t* __restrict__ w2t) {
    int idx = blockIdx.x * blockDim.x + threadIdx.x;       // idx over 1024*256
    int f = idx >> 8, c = idx & 255;
    w2t[c * HIDn + f] = f2bf(w2[f * Cn + c]);
}

// ---------------------------------------------------------------------------
// Fused: 7x7 depthwise conv + bias + NHWC permute + LayerNorm -> bf16 A matrix
// One workgroup (256 threads = 8 waves) per (b,h). Conv: thread t = channel t.
// LN: one wave per spatial position, wave32 shuffle reduction over 256 channels.
// x (103MB) is fully L2-resident (192MB), so the 7x halo reuse costs no HBM.
// ---------------------------------------------------------------------------
__global__ void conv_ln_kernel(const float* __restrict__ x,
                               const float* __restrict__ conv_w,
                               const float* __restrict__ conv_b,
                               const float* __restrict__ ln_g,
                               const float* __restrict__ ln_b,
                               uint16_t* __restrict__ A_out) {
    extern __shared__ float y_lds[];                       // [56][256]
    const int bh = blockIdx.x;
    const int b  = bh / Hn, h = bh % Hn;
    const int c  = threadIdx.x;

    float wgt[49];
#pragma unroll
    for (int i = 0; i < 49; ++i) wgt[i] = conv_w[c * 49 + i];
    const float bias = conv_b[c];
    const float* xb = x + ((size_t)b * Cn + c) * (Hn * Wn);

    for (int w = 0; w < Wn; ++w) {
        float acc = bias;
#pragma unroll
        for (int kh = 0; kh < 7; ++kh) {
            int hh = h + kh - 3;
            if (hh < 0 || hh >= Hn) continue;
            const float* row = xb + hh * Wn;
#pragma unroll
            for (int kw = 0; kw < 7; ++kw) {
                int ww = w + kw - 3;
                if (ww >= 0 && ww < Wn) acc += wgt[kh * 7 + kw] * row[ww];
            }
        }
        y_lds[w * Cn + c] = acc;
    }
    __syncthreads();

    const int wv = threadIdx.x >> 5, lane = threadIdx.x & 31;
    for (int w = wv; w < Wn; w += 8) {
        float vals[8], s = 0.f, s2 = 0.f;
#pragma unroll
        for (int j = 0; j < 8; ++j) {
            float v = y_lds[w * Cn + lane + j * 32];
            vals[j] = v; s += v; s2 += v * v;
        }
#pragma unroll
        for (int off = 16; off >= 1; off >>= 1) {          // wave32 tree reduce
            s  += __shfl_xor(s,  off);
            s2 += __shfl_xor(s2, off);
        }
        const float mu = s * (1.f / Cn);
        const float var = s2 * (1.f / Cn) - mu * mu;
        const float rs = rsqrtf(var + 1e-6f);
        const size_t rowi = (size_t)b * (Hn * Wn) + h * Wn + w;
#pragma unroll
        for (int j = 0; j < 8; ++j) {
            int ch = lane + j * 32;
            float o = (vals[j] - mu) * rs * ln_g[ch] + ln_b[ch];
            A_out[rowi * Cn + ch] = f2bf(o);
        }
    }
}

// ---------------------------------------------------------------------------
// Fused MLP: A[32x256]bf16 @ w1t -> GELU -> (LDS) -> @ w2t -> NCHW f32 out.
// 256 threads = 8 waves. GEMM1: each wave owns 128 cols of HID (2mt x 8kt x 8nt).
// GEMM2: each wave owns 32 cols of C (2mt x 32kt x 2nt). 256 wave-WMMAs each.
// 83KB LDS -> 3 workgroups/WGP -> 6 waves/SIMD for load-latency hiding.
// ---------------------------------------------------------------------------
__global__ void gemm_mlp_kernel(const uint16_t* __restrict__ A_g,
                                const uint16_t* __restrict__ w1t,
                                const uint16_t* __restrict__ w2t,
                                const float* __restrict__ b1,
                                const float* __restrict__ b2,
                                float* __restrict__ out) {
    extern __shared__ char smem[];
    char* a_lds = smem;
    char* h_lds = smem + A_LDS_BYTES;

    const int r0   = blockIdx.x * RT;
    const int tid  = threadIdx.x;
    const int wv   = tid >> 5;
    const int lane = tid & 31;
    const int lhalf = lane >> 4;    // 0/1: selects K half k0+16*lhalf..+15 (contiguous order)
    const int lmod  = lane & 15;    // M (A frag) or N (B/C/D frag)

    // Stage A tile [32][256] bf16 into padded LDS (each thread copies 64B)
    {
        const int row = tid >> 3, q = tid & 7;
        const uint4* src = (const uint4*)((const char*)A_g + ((size_t)(r0 + row) * Cn) * 2 + q * 64);
        uint4* dst = (uint4*)(a_lds + row * A_STRIDE + q * 64);
#pragma unroll
        for (int i = 0; i < 4; ++i) dst[i] = src[i];
    }
    __syncthreads();

    // ---------------- GEMM1 + bias + exact GELU -> H in LDS -----------------
    const int n0w = wv * 128;
    for (int mt = 0; mt < 2; ++mt) {
        v8f acc[8];
#pragma unroll
        for (int nt = 0; nt < 8; ++nt) acc[nt] = (v8f){0.f,0.f,0.f,0.f,0.f,0.f,0.f,0.f};

        for (int kt = 0; kt < 8; ++kt) {
            const int k0 = kt * 32;
            const char* ap = a_lds + (mt * 16 + lmod) * A_STRIDE + (k0 + lhalf * 16) * 2;
            v16bf av = frag_from(*(const uint4*)ap, *(const uint4*)(ap + 16));
#pragma unroll
            for (int nt = 0; nt < 8; ++nt) {
                const int f = n0w + nt * 16 + lmod;
                const char* bp = (const char*)w1t + ((size_t)f * Cn + k0 + lhalf * 16) * 2;
                v16bf bv = frag_from(*(const uint4*)bp, *(const uint4*)(bp + 16));
                acc[nt] = wmma_bf16(av, bv, acc[nt]);
            }
        }
#pragma unroll
        for (int nt = 0; nt < 8; ++nt) {
            const int col = n0w + nt * 16 + lmod;
            const float b1v = b1[col];
#pragma unroll
            for (int r = 0; r < 8; ++r) {
                const int row = mt * 16 + lhalf * 8 + r;
                float xv = acc[nt][r] + b1v;
                float g  = 0.5f * xv * (1.f + erff(xv * 0.70710678118654752f));
                *(uint16_t*)(h_lds + row * H_STRIDE + col * 2) = f2bf(g);
            }
        }
    }
    __syncthreads();

    // ---------------- GEMM2 + bias -> NCHW f32 output -----------------------
    const int c0w = wv * 32;
    for (int mt = 0; mt < 2; ++mt) {
        v8f acc2[2];
        acc2[0] = (v8f){0.f,0.f,0.f,0.f,0.f,0.f,0.f,0.f};
        acc2[1] = (v8f){0.f,0.f,0.f,0.f,0.f,0.f,0.f,0.f};

        for (int kt = 0; kt < 32; ++kt) {
            const int k0 = kt * 32;
            const char* ap = h_lds + (mt * 16 + lmod) * H_STRIDE + (k0 + lhalf * 16) * 2;
            v16bf av = frag_from(*(const uint4*)ap, *(const uint4*)(ap + 16));
#pragma unroll
            for (int nt = 0; nt < 2; ++nt) {
                const int cc = c0w + nt * 16 + lmod;
                const char* bp = (const char*)w2t + ((size_t)cc * HIDn + k0 + lhalf * 16) * 2;
                v16bf bv = frag_from(*(const uint4*)bp, *(const uint4*)(bp + 16));
                acc2[nt] = wmma_bf16(av, bv, acc2[nt]);
            }
        }
#pragma unroll
        for (int nt = 0; nt < 2; ++nt) {
            const int cc  = c0w + nt * 16 + lmod;
            const float b2v = b2[cc];
#pragma unroll
            for (int r = 0; r < 8; ++r) {
                const int row = r0 + mt * 16 + lhalf * 8 + r;
                const int b  = row / (Hn * Wn);
                const int hw = row - b * (Hn * Wn);
                out[((size_t)b * Cn + cc) * (Hn * Wn) + hw] = acc2[nt][r] + b2v;
            }
        }
    }
}

// ---------------------------------------------------------------------------
extern "C" void kernel_launch(void* const* d_in, const int* in_sizes, int n_in,
                              void* d_out, int out_size, void* d_ws, size_t ws_size,
                              hipStream_t stream) {
    const float* x      = (const float*)d_in[0];
    const float* conv_w = (const float*)d_in[1];
    const float* conv_b = (const float*)d_in[2];
    const float* ln_g   = (const float*)d_in[3];
    const float* ln_b   = (const float*)d_in[4];
    const float* w1     = (const float*)d_in[5];
    const float* b1     = (const float*)d_in[6];
    const float* w2     = (const float*)d_in[7];
    const float* b2     = (const float*)d_in[8];
    float* out          = (float*)d_out;

    uint16_t* A_ws = (uint16_t*)((char*)d_ws + WS_A);
    uint16_t* w1t  = (uint16_t*)((char*)d_ws + WS_W1T);
    uint16_t* w2t  = (uint16_t*)((char*)d_ws + WS_W2T);

    (void)hipFuncSetAttribute((const void*)gemm_mlp_kernel,
                              hipFuncAttributeMaxDynamicSharedMemorySize, GEMM_LDS);
    (void)hipFuncSetAttribute((const void*)conv_ln_kernel,
                              hipFuncAttributeMaxDynamicSharedMemorySize, Wn * Cn * 4);

    prep_w1t_kernel<<<(Cn * HIDn) / 256, 256, 0, stream>>>(w1, w1t);
    prep_w2t_kernel<<<(HIDn * Cn) / 256, 256, 0, stream>>>(w2, w2t);
    conv_ln_kernel<<<Bn * Hn, 256, Wn * Cn * 4, stream>>>(x, conv_w, conv_b, ln_g, ln_b, A_ws);
    gemm_mlp_kernel<<<NBLK_GEMM, 256, GEMM_LDS, stream>>>(A_ws, w1t, w2t, b1, b2, out);
}
